// AffectChannel_1185410974133
// MI455X (gfx1250) — matrix operand order repaired and checked by reading, not compiled, not run on hardware
//
#include <hip/hip_runtime.h>
#include <math.h>

typedef __bf16 bf16_t;
typedef __attribute__((ext_vector_type(16))) __bf16 v16bf;
typedef __attribute__((ext_vector_type(8)))  __bf16 v8bf;
typedef __attribute__((ext_vector_type(8)))  float  v8f;

#define D_DIM 2048
#define C_DIM 64
#define H_DIM 16
#define G_DIM 48
#define S_LEN 4096
#define B_SZ  8
#define M_TOT (B_SZ * S_LEN)   // 32768 rows

// ---- helpers -------------------------------------------------------------

__device__ __forceinline__ v16bf pack2(v8bf lo, v8bf hi) {
  v16bf r;
#pragma unroll
  for (int i = 0; i < 8; ++i) { r[i] = lo[i]; r[i + 8] = hi[i]; }
  return r;
}

__device__ __forceinline__ v16bf packA(const float4& a0, const float4& a1,
                                       const float4& a2, const float4& a3) {
  v16bf r;
  r[0]  = (bf16_t)a0.x; r[1]  = (bf16_t)a0.y; r[2]  = (bf16_t)a0.z; r[3]  = (bf16_t)a0.w;
  r[4]  = (bf16_t)a1.x; r[5]  = (bf16_t)a1.y; r[6]  = (bf16_t)a1.z; r[7]  = (bf16_t)a1.w;
  r[8]  = (bf16_t)a2.x; r[9]  = (bf16_t)a2.y; r[10] = (bf16_t)a2.z; r[11] = (bf16_t)a2.w;
  r[12] = (bf16_t)a3.x; r[13] = (bf16_t)a3.y; r[14] = (bf16_t)a3.z; r[15] = (bf16_t)a3.w;
  return r;
}

__device__ __forceinline__ void load_a4(const float* __restrict__ arow, int ka,
                                        float4& a0, float4& a1, float4& a2, float4& a3) {
  const float4* p  = (const float4*)(arow + ka);
  const float4* p2 = (const float4*)(arow + ka + 16);
  a0 = p[0]; a1 = p[1]; a2 = p2[0]; a3 = p2[1];
}

// ---- kernel 0: convert weight matrices f32 -> bf16 (tiny, one shot) ------

__global__ void cvt_weights(const float* __restrict__ cw, const float* __restrict__ wih,
                            bf16_t* __restrict__ cwb, bf16_t* __restrict__ wihb) {
  const int i = blockIdx.x * blockDim.x + threadIdx.x;
  if (i < C_DIM * D_DIM) cwb[i]  = (bf16_t)cw[i];
  if (i < G_DIM * C_DIM) wihb[i] = (bf16_t)wih[i];
}

// ---- kernel 1: compressed = gelu(residual @ compress_w^T + b), bf16 WMMA -
// One wave owns a 16x64 output tile; K = 2048 swept in bf16 k-steps of 32.
// Residual (f32) is read once and converted to bf16 in registers.
// k-loop is software-pipelined (prefetch depth 1) and unrolled 3x (63 = 21*3)
// so copy-propagation removes the double-buffer rotation movs inside the
// unrolled body.

__global__ __launch_bounds__(256) void gemm1_compress(
    const float* __restrict__ resid, const bf16_t* __restrict__ cwb,
    const float* __restrict__ cb, bf16_t* __restrict__ comp) {
  const int  lane = threadIdx.x & 31;
  const int  wave = threadIdx.x >> 5;
  const long tile = (long)blockIdx.x * 8 + wave;   // 0..2047
  const long m0   = tile * 16;
  const int  row  = lane & 15;
  const int  hf   = lane >> 4;

  v8f acc[4];
#pragma unroll
  for (int t = 0; t < 4; ++t) acc[t] = (v8f){};

  const float*  arow = resid + (m0 + row) * (long)D_DIM;
  const bf16_t* br[4];
#pragma unroll
  for (int t = 0; t < 4; ++t) br[t] = cwb + (long)(t * 16 + row) * D_DIM;

  // ---- prologue: load k0 = 0 ----
  float4 ca0, ca1, ca2, ca3;
  v8bf   cbb[8];
  load_a4(arow, hf * 8, ca0, ca1, ca2, ca3);
#pragma unroll
  for (int t = 0; t < 4; ++t) {
    cbb[2 * t]     = *(const v8bf*)(br[t] + hf * 8);
    cbb[2 * t + 1] = *(const v8bf*)(br[t] + hf * 8 + 16);
  }

  // ---- steady state: prefetch k0+32, compute k0 (63 iterations) ----
#pragma unroll 3
  for (int k0 = 0; k0 < D_DIM - 32; k0 += 32) {
    const int kan = k0 + 32 + hf * 8;
    const int kbn = k0 + 32 + hf * 8;
    float4 na0, na1, na2, na3;
    v8bf   nbb[8];
    load_a4(arow, kan, na0, na1, na2, na3);
#pragma unroll
    for (int t = 0; t < 4; ++t) {
      nbb[2 * t]     = *(const v8bf*)(br[t] + kbn);
      nbb[2 * t + 1] = *(const v8bf*)(br[t] + kbn + 16);
    }

    const v16bf af = packA(ca0, ca1, ca2, ca3);
#pragma unroll
    for (int t = 0; t < 4; ++t) {
      const v16bf bfr = pack2(cbb[2 * t], cbb[2 * t + 1]);
      acc[t] = __builtin_amdgcn_wmma_f32_16x16x32_bf16(
          false, af, false, bfr, (short)0, acc[t], false, false);
    }

    ca0 = na0; ca1 = na1; ca2 = na2; ca3 = na3;
#pragma unroll
    for (int i = 0; i < 8; ++i) cbb[i] = nbb[i];
  }

  // ---- epilogue of pipeline: compute last k-step ----
  {
    const v16bf af = packA(ca0, ca1, ca2, ca3);
#pragma unroll
    for (int t = 0; t < 4; ++t) {
      const v16bf bfr = pack2(cbb[2 * t], cbb[2 * t + 1]);
      acc[t] = __builtin_amdgcn_wmma_f32_16x16x32_bf16(
          false, af, false, bfr, (short)0, acc[t], false, false);
    }
  }

  // ---- bias + exact gelu + bf16 store ----
  const float inv_sqrt2 = 0.70710678118654752f;
#pragma unroll
  for (int t = 0; t < 4; ++t) {
    const int   c  = t * 16 + row;
    const float bb = cb[c];
#pragma unroll
    for (int r = 0; r < 8; ++r) {
      const long m = m0 + r + hf * 8;
      float v = acc[t][r] + bb;
      v = 0.5f * v * (1.0f + erff(v * inv_sqrt2));   // exact gelu
      comp[m * C_DIM + c] = (bf16_t)v;
    }
  }
}

// ---- kernel 2: x_gates = compressed @ w_ih^T + b_ih, bf16 WMMA -----------
// One wave owns a 16x48 tile; K=64 -> two k-steps; 3 N-tiles.

__global__ __launch_bounds__(256) void gemm2_gates(
    const bf16_t* __restrict__ comp, const bf16_t* __restrict__ wihb,
    const float* __restrict__ bih, float* __restrict__ xg) {
  const int  lane = threadIdx.x & 31;
  const int  wave = threadIdx.x >> 5;
  const long tile = (long)blockIdx.x * 8 + wave;
  const long m0   = tile * 16;
  const int  row  = lane & 15;
  const int  hf   = lane >> 4;

  v8f acc[3];
#pragma unroll
  for (int t = 0; t < 3; ++t) acc[t] = (v8f){};

  const bf16_t* arow = comp + (m0 + row) * (long)C_DIM;
  const bf16_t* br[3];
#pragma unroll
  for (int t = 0; t < 3; ++t) br[t] = wihb + (long)(t * 16 + row) * C_DIM;

#pragma unroll
  for (int k0 = 0; k0 < C_DIM; k0 += 32) {
    const int ka = k0 + hf * 8;
    const int kb = k0 + hf * 8;
    const v16bf af = pack2(*(const v8bf*)(arow + ka),
                           *(const v8bf*)(arow + ka + 16));
#pragma unroll
    for (int t = 0; t < 3; ++t) {
      const v16bf bfr = pack2(*(const v8bf*)(br[t] + kb),
                              *(const v8bf*)(br[t] + kb + 16));
      acc[t] = __builtin_amdgcn_wmma_f32_16x16x32_bf16(
          false, af, false, bfr, (short)0, acc[t], false, false);
    }
  }

#pragma unroll
  for (int t = 0; t < 3; ++t) {
    const int   c  = t * 16 + row;
    const float bb = bih[c];
#pragma unroll
    for (int r = 0; r < 8; ++r) {
      const long m = m0 + r + hf * 8;
      xg[m * G_DIM + c] = acc[t][r] + bb;
    }
  }
}

// ---- kernel 3: sequential GRU scan, one wave per batch -------------------
// Lane j owns hidden unit j; w_hh rows kept in registers; hidden state
// replicated per-lane and refreshed with 16 shuffles per step. Lanes 16-31
// duplicate lanes 0-15 (keeps EXEC full, only lanes<16 store).

__global__ __launch_bounds__(256) void gru_scan(
    const float* __restrict__ xg, const float* __restrict__ whh,
    const float* __restrict__ bhh, float* __restrict__ out) {
  const int lane = threadIdx.x & 31;
  const int b    = threadIdx.x >> 5;     // batch 0..7
  const int j    = lane & 15;

  float wr[16], wz[16], wn[16];
#pragma unroll
  for (int i = 0; i < 16; ++i) {
    wr[i] = whh[(0 * 16 + j) * 16 + i];
    wz[i] = whh[(1 * 16 + j) * 16 + i];
    wn[i] = whh[(2 * 16 + j) * 16 + i];
  }
  const float br = bhh[j], bz = bhh[16 + j], bn = bhh[32 + j];

  float h[16];
#pragma unroll
  for (int i = 0; i < 16; ++i) h[i] = 0.0f;
  float hj = 0.0f;                       // this lane's own previous h[j]

  const float* xrow = xg  + (long)b * S_LEN * G_DIM;
  float*       orow = out + (long)b * S_LEN * H_DIM;

  for (int s = 0; s < S_LEN; ++s) {
    const float xr = xrow[j], xz = xrow[16 + j], xn = xrow[32 + j];
    float ar = br, az = bz, an = bn;
#pragma unroll
    for (int i = 0; i < 16; ++i) {
      ar = fmaf(wr[i], h[i], ar);
      az = fmaf(wz[i], h[i], az);
      an = fmaf(wn[i], h[i], an);
    }
    const float r = 1.0f / (1.0f + __expf(-(xr + ar)));
    const float z = 1.0f / (1.0f + __expf(-(xz + az)));
    const float n = tanhf(xn + r * an);
    const float hn = (1.0f - z) * n + z * hj;

    if (lane < 16) orow[s * H_DIM + j] = hn;

#pragma unroll
    for (int i = 0; i < 16; ++i) h[i] = __shfl(hn, i, 32);
    hj = hn;
    xrow += G_DIM;
  }
}

// ---- launch --------------------------------------------------------------

extern "C" void kernel_launch(void* const* d_in, const int* in_sizes, int n_in,
                              void* d_out, int out_size, void* d_ws, size_t ws_size,
                              hipStream_t stream) {
  const float* residual   = (const float*)d_in[0];
  const float* compress_w = (const float*)d_in[1];
  const float* compress_b = (const float*)d_in[2];
  const float* w_ih       = (const float*)d_in[3];
  const float* w_hh       = (const float*)d_in[4];
  const float* b_ih       = (const float*)d_in[5];
  const float* b_hh       = (const float*)d_in[6];
  float* out = (float*)d_out;

  char* ws = (char*)d_ws;
  bf16_t* cwb  = (bf16_t*)(ws + 0);         //  64*2048*2 =  262144 B
  bf16_t* wihb = (bf16_t*)(ws + 262144);    //  48*64*2   =    6144 B
  bf16_t* comp = (bf16_t*)(ws + 268288);    //  32768*64*2 = 4194304 B
  float*  xg   = (float*) (ws + 4462592);   //  32768*48*4 = 6291456 B  (total ~10.3 MB)

  hipLaunchKernelGGL(cvt_weights, dim3(512), dim3(256), 0, stream,
                     compress_w, w_ih, cwb, wihb);
  hipLaunchKernelGGL(gemm1_compress, dim3(M_TOT / 16 / 8), dim3(256), 0, stream,
                     residual, cwb, compress_b, comp);
  hipLaunchKernelGGL(gemm2_gates, dim3(M_TOT / 16 / 8), dim3(256), 0, stream,
                     comp, wihb, b_ih, xg);
  hipLaunchKernelGGL(gru_scan, dim3(1), dim3(256), 0, stream,
                     xg, w_hh, b_hh, out);
}
